// ET_test_33071248179241
// MI455X (gfx1250) — matrix operand order repaired and checked by reading, not compiled, not run on hardware
//
#include <hip/hip_runtime.h>
#include <hip/hip_bf16.h>
#include <math.h>

typedef __attribute__((ext_vector_type(16))) _Float16 v16h;
typedef __attribute__((ext_vector_type(8)))  float    v8f;

#define BSZ  512
#define NR_  60
#define NK_  13
#define NRB  30720   // NR_ * BSZ

// ------------------------------------------------------------------
// Repack fp32 weights (Cout,Cin,NK) -> f16 WMMA B-fragment order:
//   out[(((kb*(Cout/16) + nb)*32 + lane)*16) + h]
//     = W[o = nb*16 + lane%16][c][kcoef],  K = kb*32 + (h<8 ? h : h+8) + (lane>=16 ? 8:0)
//     with K = kcoef*Cin + c   (kcoef outer, c inner; Cin % 32 == 0)
// ------------------------------------------------------------------
__global__ void so3_prep_weight(const float* __restrict__ W,
                                _Float16* __restrict__ out,
                                int Cin, int Cout, int NKp, int total) {
    int idx = blockIdx.x * blockDim.x + threadIdx.x;
    if (idx >= total) return;
    int h    = idx & 15;
    int lane = (idx >> 4) & 31;
    int t2   = idx >> 9;              // kb * (Cout/16) + nb
    int nblk = Cout >> 4;
    int kb   = t2 / nblk;
    int nb   = t2 - kb * nblk;
    int kk   = ((h < 8) ? h : h + 8) + ((lane >= 16) ? 8 : 0);
    int K    = kb * 32 + kk;
    int n    = nb * 16 + (lane & 15);
    int kco  = K / Cin;
    int c    = K - kco * Cin;
    out[idx] = (_Float16)W[(size_t)n * Cin * NKp + (size_t)c * NKp + kco];
}

// ------------------------------------------------------------------
// Build feats (128ch) in f16 [M = r*B + b][C] layout, applying permutation
// ------------------------------------------------------------------
__global__ void so3_feats(const float* __restrict__ be0, const float* __restrict__ be1,
                          const float* __restrict__ af0, const float* __restrict__ af1,
                          const int* __restrict__ pre_idx, const int* __restrict__ permu,
                          _Float16* __restrict__ out) {
    int idx = blockIdx.x * blockDim.x + threadIdx.x;
    if (idx >= 128 * NRB) return;
    int c = idx & 127;
    int m = idx >> 7;
    int r = m >> 9;          // m / BSZ
    int b = m & 511;         // m % BSZ
    float v;
    if (c < 32) {
        int rr = permu[pre_idx[b] * NR_ + r];
        v = be0[(size_t)b * 32 * NR_ + (size_t)c * NR_ + rr];
    } else if (c < 64) {
        v = be1[(size_t)b * 32 * NR_ + (size_t)(c - 32) * NR_ + r];
    } else if (c < 96) {
        int rr = permu[pre_idx[b] * NR_ + r];
        v = af0[(size_t)b * 32 * NR_ + (size_t)(c - 64) * NR_ + rr];
    } else {
        v = af1[(size_t)b * 32 * NR_ + (size_t)(c - 96) * NR_ + r];
    }
    out[idx] = (_Float16)v;
}

// ------------------------------------------------------------------
// LDS-free gather-GEMM, v_wmma_f32_16x16x32_f16, 2x2 register blocking.
//   Y[(r*B + b)][o] = sum_{kcoef,c} Aact[nei[r,kcoef]*B + b][c] * W[o][c][kcoef] + bias[o]
// Block = 128 threads (4 waves); block tile 64(M) x 64(N); wave tile 32x32.
// A frag: two b128 loads per lane (activation rows contiguous in C).
// B frag: two b128 loads per lane from prepacked fragment-order weights.
// ------------------------------------------------------------------
__global__ __launch_bounds__(128)
void so3_gemm_wmma(const _Float16* __restrict__ Aact,
                   const _Float16* __restrict__ Bwf,
                   const float* __restrict__ bias,
                   const int* __restrict__ nei,
                   int NKp, int Cin, int Cout,
                   float* __restrict__ Y) {
    const int r    = blockIdx.x;
    const int b0   = blockIdx.y * 64;
    const int n0   = blockIdx.z * 64;
    const int tid  = threadIdx.x;
    const int lane = tid & 31;
    const int wave = tid >> 5;
    const int mw   = wave & 1;            // 32-row half of block tile
    const int nw   = wave >> 1;           // 32-col half of block tile
    const int l15  = lane & 15;
    const int lhi  = (lane >= 16) ? 8 : 0;

    const int bgA0 = b0 + mw * 32 + l15;  // batch row of A-frag 0 for this lane
    const int bgA1 = bgA0 + 16;           // A-frag 1
    const int nb0  = (n0 >> 4) + nw * 2;  // 16-wide N blocks for B frags
    const size_t bOff0 = ((size_t)(nb0)     * 32 + lane) << 4;  // halves
    const size_t bOff1 = ((size_t)(nb0 + 1) * 32 + lane) << 4;
    const size_t bStride = (size_t)Cout * 32;                   // halves per 32-K block

    union { v8f v; float e[8]; } acc00, acc01, acc10, acc11;
#pragma unroll
    for (int i = 0; i < 8; ++i) {
        acc00.e[i] = 0.f; acc01.e[i] = 0.f; acc10.e[i] = 0.f; acc11.e[i] = 0.f;
    }

    int kb = 0;
    for (int kcoef = 0; kcoef < NKp; ++kcoef) {
        const int pos = (NKp == 1) ? r : nei[r * NKp + kcoef];
        const _Float16* rowA0 = Aact + ((size_t)pos * BSZ + bgA0) * Cin + lhi;
        const _Float16* rowA1 = Aact + ((size_t)pos * BSZ + bgA1) * Cin + lhi;
        for (int c0 = 0; c0 < Cin; c0 += 32, ++kb) {
            union { v16h v; uint4 q[2]; } a0, a1, w0, w1;
            // A 16x32 f16 frag: halves 0..7 = K c0+lhi..+7 ; halves 8..15 = K c0+lhi+16..+23
            a0.q[0] = *(const uint4*)(rowA0 + c0);
            a0.q[1] = *(const uint4*)(rowA0 + c0 + 16);
            a1.q[0] = *(const uint4*)(rowA1 + c0);
            a1.q[1] = *(const uint4*)(rowA1 + c0 + 16);
            // B 32x16 f16 frags: prepacked, 32B contiguous per lane
            const _Float16* pb = Bwf + (size_t)kb * bStride;
            w0.q[0] = *(const uint4*)(pb + bOff0);
            w0.q[1] = *(const uint4*)(pb + bOff0 + 8);
            w1.q[0] = *(const uint4*)(pb + bOff1);
            w1.q[1] = *(const uint4*)(pb + bOff1 + 8);

            acc00.v = __builtin_amdgcn_wmma_f32_16x16x32_f16(false, a0.v, false, w0.v, (short)0, acc00.v, false, false);
            acc01.v = __builtin_amdgcn_wmma_f32_16x16x32_f16(false, a0.v, false, w1.v, (short)0, acc01.v, false, false);
            acc10.v = __builtin_amdgcn_wmma_f32_16x16x32_f16(false, a1.v, false, w0.v, (short)0, acc10.v, false, false);
            acc11.v = __builtin_amdgcn_wmma_f32_16x16x32_f16(false, a1.v, false, w1.v, (short)0, acc11.v, false, false);
        }
    }

    // C/D f32 16x16 layout: VGPR vi -> M = vi (lanes 0-15) / vi+8 (lanes 16-31), N = lane%16
    const int og0 = n0 + nw * 32 + l15;
    const int og1 = og0 + 16;
    const float bias0 = bias[og0];
    const float bias1 = bias[og1];
    const int mBase = b0 + mw * 32 + ((lane >= 16) ? 8 : 0);
#pragma unroll
    for (int vi = 0; vi < 8; ++vi) {
        size_t rowA = ((size_t)r * BSZ + (mBase + vi)) * Cout;
        size_t rowB = ((size_t)r * BSZ + (mBase + 16 + vi)) * Cout;
        Y[rowA + og0] = acc00.e[vi] + bias0;
        Y[rowA + og1] = acc01.e[vi] + bias1;
        Y[rowB + og0] = acc10.e[vi] + bias0;
        Y[rowB + og1] = acc11.e[vi] + bias1;
    }
}

// ------------------------------------------------------------------
// Per-channel sum / sum-of-squares over M=NRB rows of [M][C] tensor
// ------------------------------------------------------------------
__global__ void so3_stats(const float* __restrict__ Y, int C,
                          float* __restrict__ sums, float* __restrict__ sqs) {
    __shared__ float s1[256], s2[256];
    int c = blockIdx.x, t = threadIdx.x;
    float a = 0.f, b = 0.f;
    for (int m = t; m < NRB; m += 256) {
        float v = Y[(size_t)m * C + c];
        a += v; b += v * v;
    }
    s1[t] = a; s2[t] = b;
    __syncthreads();
    for (int o = 128; o > 0; o >>= 1) {
        if (t < o) { s1[t] += s1[t + o]; s2[t] += s2[t + o]; }
        __syncthreads();
    }
    if (t == 0) { sums[c] = s1[0]; sqs[c] = s2[0]; }
}

__device__ __forceinline__ float bn_apply(float v, float sum, float sq,
                                          float g, float be) {
    const float inv = 1.0f / (float)NRB;
    float m   = sum * inv;
    float var = sq * inv - m * m;
    float s   = g * rsqrtf(var + 1e-5f);
    return (v - m) * s + be;
}

// BN + ReLU -> f16 ([M][C] in, [M][C] out; C is a power of two)
__global__ void so3_bnrelu_f16(const float* __restrict__ Y,
                               const float* __restrict__ sums, const float* __restrict__ sqs,
                               const float* __restrict__ g, const float* __restrict__ be,
                               _Float16* __restrict__ out, int Cmask, int total) {
    int idx = blockIdx.x * blockDim.x + threadIdx.x;
    if (idx >= total) return;
    int c = idx & Cmask;
    float v = bn_apply(Y[idx], sums[c], sqs[c], g[c], be[c]);
    v = v > 0.f ? v : 0.f;
    out[idx] = (_Float16)v;
}

// x2 = relu( relu(bn3(y3)) + bn_sc(ysc) ) -> f16
__global__ void so3_combine_f16(const float* __restrict__ y3,
                                const float* __restrict__ s3, const float* __restrict__ q3,
                                const float* __restrict__ g3, const float* __restrict__ b3,
                                const float* __restrict__ ysc,
                                const float* __restrict__ ss, const float* __restrict__ qs,
                                const float* __restrict__ gs, const float* __restrict__ bs,
                                _Float16* __restrict__ out, int Cmask, int total) {
    int idx = blockIdx.x * blockDim.x + threadIdx.x;
    if (idx >= total) return;
    int c = idx & Cmask;
    float h = bn_apply(y3[idx], s3[c], q3[c], g3[c], b3[c]);
    h = h > 0.f ? h : 0.f;
    float sc = bn_apply(ysc[idx], ss[c], qs[c], gs[c], bs[c]);
    float v = h + sc;
    v = v > 0.f ? v : 0.f;
    out[idx] = (_Float16)v;
}

// fc3 at r==0 (BN+ReLU of yfc2 inline, fp32) + quaternion normalize
__global__ void so3_fc3_norm(const float* __restrict__ yfc2,   // [M][128]
                             const float* __restrict__ sums, const float* __restrict__ sqs,
                             const float* __restrict__ g, const float* __restrict__ be,
                             const float* __restrict__ W, const float* __restrict__ bb,
                             float* __restrict__ out) {
    int b = blockIdx.x * blockDim.x + threadIdx.x;
    if (b >= BSZ) return;
    float acc[4] = {bb[0], bb[1], bb[2], bb[3]};
    const float* row = yfc2 + (size_t)b * 128;   // m = 0*BSZ + b
    for (int c = 0; c < 128; ++c) {
        float v = bn_apply(row[c], sums[c], sqs[c], g[c], be[c]);
        v = v > 0.f ? v : 0.f;
#pragma unroll
        for (int o = 0; o < 4; ++o) acc[o] += v * W[o * 128 + c];
    }
    float nrm = rsqrtf(acc[0]*acc[0] + acc[1]*acc[1] + acc[2]*acc[2] + acc[3]*acc[3]);
#pragma unroll
    for (int o = 0; o < 4; ++o) out[b * 4 + o] = acc[o] * nrm;
}

// ------------------------------------------------------------------
extern "C" void kernel_launch(void* const* d_in, const int* in_sizes, int n_in,
                              void* d_out, int out_size, void* d_ws, size_t ws_size,
                              hipStream_t stream) {
    const float* be0 = (const float*)d_in[0];
    const float* be1 = (const float*)d_in[1];
    const float* af0 = (const float*)d_in[2];
    const float* af1 = (const float*)d_in[3];
    const int* pre_idx = (const int*)d_in[n_in - 3];
    const int* nei     = (const int*)d_in[n_in - 2];
    const int* permu   = (const int*)d_in[n_in - 1];

    // layer slots: 0 init, 1 res_c1, 2 res_c2, 3 sc, 4 fc1, 5 fc2, 6 fc3 (W,b only)
    int iW[7], ib[7], ig[7], ibe[7];
    if (in_sizes[4] == 425984) {
        // dict insertion order; per-layer W,b,g,beta
        int base = 4;
        for (int L = 0; L < 6; ++L) {
            iW[L] = base; ib[L] = base + 1; ig[L] = base + 2; ibe[L] = base + 3;
            base += 4;
        }
        iW[6] = base; ib[6] = base + 1;
    } else {
        // jax sorted pytree: fc1, fc2, fc3, init, res_c1, res_c2, sc ; per-layer W,b,beta,g
        int order[7] = {4, 5, 6, 0, 1, 2, 3};
        int base = 4;
        for (int k = 0; k < 7; ++k) {
            int L = order[k];
            if (L == 6) { iW[6] = base; ib[6] = base + 1; base += 2; }
            else { iW[L] = base; ib[L] = base + 1; ibe[L] = base + 2; ig[L] = base + 3; base += 4; }
        }
    }
    auto F = [&](int i) { return (const float*)d_in[i]; };

    // ---------------- workspace carve (lifetime-overlaid) ----------------
    char* ws = (char*)d_ws;
    size_t off = 0;
    auto carve = [&](size_t bytes) -> char* {
        char* p = ws + off;
        off = (off + bytes + 255) & ~(size_t)255;
        return p;
    };
    _Float16* W1  = (_Float16*)carve((size_t)1664 * 256 * 2);
    _Float16* W2  = (_Float16*)carve((size_t)3328 * 512 * 2);
    _Float16* W3  = (_Float16*)carve((size_t)6656 * 256 * 2);
    _Float16* Wsc = (_Float16*)carve((size_t)256 * 256 * 2);
    _Float16* Wf1 = (_Float16*)carve((size_t)256 * 512 * 2);
    _Float16* Wf2 = (_Float16*)carve((size_t)512 * 128 * 2);
    float*    st  = (float*)carve((size_t)6 * 1024 * 4);     // slot*1024: sum, +512: sumsq
    _Float16* feats = (_Float16*)carve((size_t)128 * NRB * 2);
    float*    P1  = (float*)carve((size_t)512 * NRB * 4);    // raw GEMM outputs
    _Float16* P2  = (_Float16*)carve((size_t)512 * NRB * 2); // f16 activations
    float*    P3  = (float*)carve((size_t)256 * NRB * 4);    // ysc then yfc2
    auto SUM = [&](int s) { return st + s * 1024; };
    auto SQ  = [&](int s) { return st + s * 1024 + 512; };

    // ---------------- weight repack (once per call) ----------------
    auto prep = [&](const float* W, _Float16* outp, int Cin, int Cout, int NKp) {
        int total = Cin * NKp * Cout;
        so3_prep_weight<<<(total + 255) / 256, 256, 0, stream>>>(W, outp, Cin, Cout, NKp, total);
    };
    prep(F(iW[0]), W1,  128, 256, 13);
    prep(F(iW[1]), W2,  256, 512, 13);
    prep(F(iW[2]), W3,  512, 256, 13);
    prep(F(iW[3]), Wsc, 256, 256, 1);
    prep(F(iW[4]), Wf1, 256, 512, 1);
    prep(F(iW[5]), Wf2, 512, 128, 1);

    so3_feats<<<(128 * NRB + 255) / 256, 256, 0, stream>>>(be0, be1, af0, af1,
                                                           pre_idx, permu, feats);

    // init conv: 128 -> 256
    float* y1 = P1;
    so3_gemm_wmma<<<dim3(NR_, 8, 4), 128, 0, stream>>>(feats, W1, F(ib[0]), nei, 13, 128, 256, y1);
    so3_stats<<<256, 256, 0, stream>>>(y1, 256, SUM(0), SQ(0));
    _Float16* x16 = P2;
    so3_bnrelu_f16<<<(256 * NRB + 255) / 256, 256, 0, stream>>>(
        y1, SUM(0), SQ(0), F(ig[0]), F(ibe[0]), x16, 255, 256 * NRB);

    // shortcut fc (raw, BN later) and res_c1 both consume x16
    float* ysc = P3;
    so3_gemm_wmma<<<dim3(NR_, 8, 4), 128, 0, stream>>>(x16, Wsc, F(ib[3]), nei, 1, 256, 256, ysc);
    float* y2 = P1;
    so3_gemm_wmma<<<dim3(NR_, 8, 8), 128, 0, stream>>>(x16, W2, F(ib[1]), nei, 13, 256, 512, y2);
    so3_stats<<<512, 256, 0, stream>>>(y2, 512, SUM(1), SQ(1));
    _Float16* h16 = P2;  // x16 dead
    so3_bnrelu_f16<<<(512 * NRB + 255) / 256, 256, 0, stream>>>(
        y2, SUM(1), SQ(1), F(ig[1]), F(ibe[1]), h16, 511, 512 * NRB);

    // res_c2: 512 -> 256
    float* y3 = P1;  // y2 dead
    so3_gemm_wmma<<<dim3(NR_, 8, 4), 128, 0, stream>>>(h16, W3, F(ib[2]), nei, 13, 512, 256, y3);
    so3_stats<<<256, 256, 0, stream>>>(y3, 256, SUM(2), SQ(2));
    so3_stats<<<256, 256, 0, stream>>>(ysc, 256, SUM(3), SQ(3));
    _Float16* x2 = P2;  // h16 dead
    so3_combine_f16<<<(256 * NRB + 255) / 256, 256, 0, stream>>>(
        y3, SUM(2), SQ(2), F(ig[2]), F(ibe[2]),
        ysc, SUM(3), SQ(3), F(ig[3]), F(ibe[3]), x2, 255, 256 * NRB);

    // fc1: 256 -> 512
    float* yf1 = P1;  // y3 dead
    so3_gemm_wmma<<<dim3(NR_, 8, 8), 128, 0, stream>>>(x2, Wf1, F(ib[4]), nei, 1, 256, 512, yf1);
    so3_stats<<<512, 256, 0, stream>>>(yf1, 512, SUM(4), SQ(4));
    _Float16* a1 = P2;  // x2 dead
    so3_bnrelu_f16<<<(512 * NRB + 255) / 256, 256, 0, stream>>>(
        yf1, SUM(4), SQ(4), F(ig[4]), F(ibe[4]), a1, 511, 512 * NRB);

    // fc2: 512 -> 128
    float* yf2 = P3;  // ysc dead
    so3_gemm_wmma<<<dim3(NR_, 8, 2), 128, 0, stream>>>(a1, Wf2, F(ib[5]), nei, 1, 512, 128, yf2);
    so3_stats<<<128, 256, 0, stream>>>(yf2, 128, SUM(5), SQ(5));

    // fc3 at r=0 + normalize
    so3_fc3_norm<<<2, 256, 0, stream>>>(yf2, SUM(5), SQ(5), F(ig[5]), F(ibe[5]),
                                        F(iW[6]), F(ib[6]), (float*)d_out);
    (void)ws_size; (void)out_size;
}
